// SimpleUpBlock_49478023250390
// MI455X (gfx1250) — matrix operand order repaired and checked by reading, not compiled, not run on hardware
//
#include <hip/hip_runtime.h>

// ---------------------------------------------------------------------------
// CDNA5 (gfx1250) wave32 WMMA implementation of the sparse-conv up-block.
// All GEMMs use v_wmma_f32_16x16x32_f16 (f16 inputs, f32 accumulate).
// Tiling: one wave owns a 16-row x 64-col output stripe (4 accumulators),
// so each A fragment (incl. the expensive rulebook gathers) is loaded once
// and feeds 4 column-tiles of WMMAs.
// ---------------------------------------------------------------------------

typedef __attribute__((ext_vector_type(16))) _Float16 v16h;
typedef __attribute__((ext_vector_type(8)))  _Float16 v8h;
typedef __attribute__((ext_vector_type(8)))  float    v8f;

#define WMMA_F16(A, B, C) \
  __builtin_amdgcn_wmma_f32_16x16x32_f16(false, (A), false, (B), (short)0, (C), false, false)

// ---------------------------------------------------------------------------
// Zero the BN statistics accumulators (ws is poisoned, not zeroed, by harness)
// ---------------------------------------------------------------------------
__global__ void k_zero(float* p, int n) {
  int i = blockIdx.x * blockDim.x + threadIdx.x;
  if (i < n) p[i] = 0.0f;
}

// ---------------------------------------------------------------------------
// Convert weights f32 -> f16, transposed so B fragments are contiguous per lane
//   wfT [64][128]    = w_fuse[k][n] transposed
//   w3T [27][64][64] = w3[k][cin][cout] -> [k][cout][cin]
//   wupT[8][64][64]  = wup[k][c][f]    -> [k][f][c]
// ---------------------------------------------------------------------------
__global__ void k_prep(const float* __restrict__ w_fuse, const float* __restrict__ w3,
                       const float* __restrict__ wup,
                       _Float16* __restrict__ wfT, _Float16* __restrict__ w3T,
                       _Float16* __restrict__ wupT) {
  int i = blockIdx.x * 256 + threadIdx.x;
  if (i < 64 * 128) {
    int n = i >> 7, k = i & 127;
    wfT[i] = (_Float16)w_fuse[k * 64 + n];
    return;
  }
  int j = i - 64 * 128;
  if (j < 27 * 4096) {
    int k = j >> 12, r = j & 4095, n = r >> 6, c = r & 63;
    w3T[j] = (_Float16)w3[k * 4096 + c * 64 + n];
    return;
  }
  int l = j - 27 * 4096;
  if (l < 8 * 4096) {
    int k = l >> 12, r = l & 4095, f = r >> 6, c = r & 63;
    wupT[l] = (_Float16)wup[k * 4096 + c * 64 + f];
  }
}

// ---------------------------------------------------------------------------
// Time embedding: emb[b][c] = sum_e silu(t[b][e]) * w_temb[e][c] + b_temb[c]
// ---------------------------------------------------------------------------
__global__ void k_emb(const float* __restrict__ t, const float* __restrict__ w_temb,
                      const float* __restrict__ b_temb, float* __restrict__ emb) {
  int b = threadIdx.x >> 6, c = threadIdx.x & 63;
  float s = b_temb[c];
  for (int e = 0; e < 256; ++e) {
    float x = t[b * 256 + e];
    float sx = x / (1.0f + __expf(-x));
    s = fmaf(sx, w_temb[e * 64 + c], s);
  }
  emb[b * 64 + c] = s;
}

// ---------------------------------------------------------------------------
// GEMM1: y[N,64] = feats[N,128] @ w_fuse  (+ per-channel sum/sumsq for BN)
// 256 thr = 8 waves; each wave: 16 rows x 64 cols, K=128 -> 16 WMMAs/wave.
// ---------------------------------------------------------------------------
__global__ __launch_bounds__(256)
void k_gemm_fuse(const float* __restrict__ feats, const _Float16* __restrict__ wfT,
                 float* __restrict__ y, float* __restrict__ stats, int N) {
  __shared__ float s_sum[64], s_sq[64];
  if (threadIdx.x < 64) { s_sum[threadIdx.x] = 0.0f; s_sq[threadIdx.x] = 0.0f; }
  __syncthreads();

  const int wv   = threadIdx.x >> 5;
  const int lane = threadIdx.x & 31;
  const int lrow = lane & 15;
  const int hs   = lane >> 4;           // K-phase select per ISA A-layout
  const int row0 = blockIdx.x * 128 + wv * 16;
  const int row  = row0 + lrow;
  const int rc   = row < N ? row : N - 1;

  v8f acc[4];
#pragma unroll
  for (int ct = 0; ct < 4; ++ct) { v8f z = {}; acc[ct] = z; }

#pragma unroll
  for (int kk = 0; kk < 4; ++kk) {
    // A fragment: lane holds row `rc`, K = kk*32 + hs*16 + {0..15}
    const v8f* ap = (const v8f*)(feats + (size_t)rc * 128 + kk * 32 + hs * 16);
    union { v16h v; v8h p[2]; } a;
    a.p[0] = __builtin_convertvector(ap[0], v8h);
    a.p[1] = __builtin_convertvector(ap[1], v8h);   // +8 floats
#pragma unroll
    for (int ct = 0; ct < 4; ++ct) {
      int col = ct * 16 + lrow;
      v16h b = *(const v16h*)(wfT + (size_t)col * 128 + kk * 32 + hs * 16);
      acc[ct] = WMMA_F16(a.v, b, acc[ct]);
    }
  }

  const int orow0 = row0 + hs * 8;      // D layout: vgpr r -> row hs*8+r
#pragma unroll
  for (int ct = 0; ct < 4; ++ct) {
    int col = ct * 16 + lrow;
    float lsum = 0.0f, lsq = 0.0f;
#pragma unroll
    for (int r = 0; r < 8; ++r) {
      int orow = orow0 + r;
      if (orow < N) {
        float v = acc[ct][r];
        y[(size_t)orow * 64 + col] = v;
        lsum += v; lsq += v * v;
      }
    }
    atomicAdd(&s_sum[col], lsum);
    atomicAdd(&s_sq[col], lsq);
  }
  __syncthreads();
  if (threadIdx.x < 64) {
    atomicAdd(&stats[threadIdx.x],      s_sum[threadIdx.x]);
    atomicAdd(&stats[64 + threadIdx.x], s_sq[threadIdx.x]);
  }
}

// ---------------------------------------------------------------------------
// BN finalize: (sum, sumsq) -> (scale, shift)
// ---------------------------------------------------------------------------
__global__ void k_finalize(const float* __restrict__ stats, const float* __restrict__ g,
                           const float* __restrict__ b, float* __restrict__ ss, float inv_n) {
  int c = threadIdx.x;
  if (c < 64) {
    float m   = stats[c] * inv_n;
    float var = fmaxf(stats[64 + c] * inv_n - m * m, 0.0f);
    float sc  = g[c] * rsqrtf(var + 1e-5f);
    ss[c]      = sc;
    ss[64 + c] = b[c] - m * sc;
  }
}

// ---------------------------------------------------------------------------
// BN1 + ReLU + time-embedding add; emit h as f16 for the WMMA conv
// ---------------------------------------------------------------------------
__global__ void k_bn1(const float* __restrict__ y, const float* __restrict__ ss,
                      const float* __restrict__ emb, const int* __restrict__ bidx,
                      _Float16* __restrict__ h, int total) {
  int i = blockIdx.x * 256 + threadIdx.x;
  if (i < total) {
    int c = i & 63, n = i >> 6;
    float v = fmaxf(fmaf(y[i], ss[c], ss[64 + c]), 0.0f) + emb[bidx[n] * 64 + c];
    h[i] = (_Float16)v;
  }
}

// ---------------------------------------------------------------------------
// 3x3x3 submanifold sparse conv: out[n] = sum_k h[nbr[k][n]] @ w3[k]
// 128 thr = 4 waves; each wave: 16 rows x 64 cols. One gathered A fragment
// feeds 8 WMMAs per live rulebook offset. Misses are lane-zeroed so EXEC is
// all-ones at every WMMA; dead offsets skipped wave-uniformly via __any.
// ---------------------------------------------------------------------------
__global__ __launch_bounds__(128)
void k_conv(const _Float16* __restrict__ h, const _Float16* __restrict__ w3T,
            const int* __restrict__ nbr, float* __restrict__ y2,
            float* __restrict__ stats, int N) {
  __shared__ float s_sum[64], s_sq[64];
  if (threadIdx.x < 64) { s_sum[threadIdx.x] = 0.0f; s_sq[threadIdx.x] = 0.0f; }
  __syncthreads();

  const int wv   = threadIdx.x >> 5;
  const int lane = threadIdx.x & 31;
  const int lrow = lane & 15;
  const int hs   = lane >> 4;
  const int row0 = blockIdx.x * 64 + wv * 16;
  const int n    = row0 + lrow;
  const int nc   = n < N ? n : N - 1;
  const bool nvalid = n < N;

  v8f acc[4];
#pragma unroll
  for (int ct = 0; ct < 4; ++ct) { v8f z = {}; acc[ct] = z; }
  const v8h z8 = {};

  for (int k = 0; k < 27; ++k) {
    int idx = nvalid ? nbr[(size_t)k * N + nc] : -1;
    if (k + 1 < 27) __builtin_prefetch(&nbr[(size_t)(k + 1) * N + nc], 0, 0);
    if (__any(idx >= 0)) {              // wave-uniform skip of dead offsets
      int ic = idx >= 0 ? idx : 0;
      const _Float16* hp = h + (size_t)ic * 64 + hs * 16;
      union { v16h v; v8h p[2]; } a0, a1;
      a0.p[0] = *(const v8h*)(hp);         // K = hs*16 + 0..7
      a0.p[1] = *(const v8h*)(hp + 8);     // K = hs*16 + 8..15
      a1.p[0] = *(const v8h*)(hp + 32);
      a1.p[1] = *(const v8h*)(hp + 40);
      if (idx < 0) { a0.p[0] = z8; a0.p[1] = z8; a1.p[0] = z8; a1.p[1] = z8; }
      const _Float16* wk = w3T + (size_t)k * 4096 + hs * 16;
#pragma unroll
      for (int ct = 0; ct < 4; ++ct) {
        const _Float16* wp = wk + (size_t)(ct * 16 + lrow) * 64;
        v16h b0 = *(const v16h*)(wp);
        v16h b1 = *(const v16h*)(wp + 32);
        acc[ct] = WMMA_F16(a0.v, b0, acc[ct]);
        acc[ct] = WMMA_F16(a1.v, b1, acc[ct]);
      }
    }
  }

  const int orow0 = row0 + hs * 8;
#pragma unroll
  for (int ct = 0; ct < 4; ++ct) {
    int col = ct * 16 + lrow;
    float lsum = 0.0f, lsq = 0.0f;
#pragma unroll
    for (int r = 0; r < 8; ++r) {
      int orow = orow0 + r;
      if (orow < N) {
        float v = acc[ct][r];
        y2[(size_t)orow * 64 + col] = v;
        lsum += v; lsq += v * v;
      }
    }
    atomicAdd(&s_sum[col], lsum);
    atomicAdd(&s_sq[col], lsq);
  }
  __syncthreads();
  if (threadIdx.x < 64) {
    atomicAdd(&stats[threadIdx.x],      s_sum[threadIdx.x]);
    atomicAdd(&stats[64 + threadIdx.x], s_sq[threadIdx.x]);
  }
}

// ---------------------------------------------------------------------------
// BN2 + ReLU -> h2 (f16)
// ---------------------------------------------------------------------------
__global__ void k_bn2(const float* __restrict__ y, const float* __restrict__ ss,
                      _Float16* __restrict__ h2, int total) {
  int i = blockIdx.x * 256 + threadIdx.x;
  if (i < total) {
    int c = i & 63;
    h2[i] = (_Float16)fmaxf(fmaf(y[i], ss[c], ss[64 + c]), 0.0f);
  }
}

// ---------------------------------------------------------------------------
// Transposed conv k=2,s=2: for each of 8 children, out[n*8+k] = h2[n] @ wup[k]
// One A fragment per wave feeds 64 WMMAs (8 children x 4 col-tiles x 2 K-steps)
// ---------------------------------------------------------------------------
__global__ __launch_bounds__(128)
void k_upconv(const _Float16* __restrict__ h2, const _Float16* __restrict__ wupT,
              float* __restrict__ out, float* __restrict__ stats, int N) {
  __shared__ float s_sum[64], s_sq[64];
  if (threadIdx.x < 64) { s_sum[threadIdx.x] = 0.0f; s_sq[threadIdx.x] = 0.0f; }
  __syncthreads();

  const int wv   = threadIdx.x >> 5;
  const int lane = threadIdx.x & 31;
  const int lrow = lane & 15;
  const int hs   = lane >> 4;
  const int row0 = blockIdx.x * 64 + wv * 16;
  const int n    = row0 + lrow;
  const int nc   = n < N ? n : N - 1;

  const _Float16* hp = h2 + (size_t)nc * 64 + hs * 16;
  union { v16h v; v8h p[2]; } a0, a1;
  a0.p[0] = *(const v8h*)(hp);
  a0.p[1] = *(const v8h*)(hp + 8);
  a1.p[0] = *(const v8h*)(hp + 32);
  a1.p[1] = *(const v8h*)(hp + 40);

  float lsum[4] = {0.0f, 0.0f, 0.0f, 0.0f};
  float lsq[4]  = {0.0f, 0.0f, 0.0f, 0.0f};
  const int orow0 = row0 + hs * 8;

#pragma unroll
  for (int k = 0; k < 8; ++k) {
#pragma unroll
    for (int ct = 0; ct < 4; ++ct) {
      int col = ct * 16 + lrow;
      const _Float16* wp = wupT + (size_t)k * 4096 + (size_t)col * 64 + hs * 16;
      v16h b0 = *(const v16h*)(wp);
      v16h b1 = *(const v16h*)(wp + 32);
      v8f acc = {};
      acc = WMMA_F16(a0.v, b0, acc);
      acc = WMMA_F16(a1.v, b1, acc);
#pragma unroll
      for (int r = 0; r < 8; ++r) {
        int nr = orow0 + r;
        if (nr < N) {
          float v = acc[r];
          out[((size_t)nr * 8 + k) * 64 + col] = v;
          lsum[ct] += v; lsq[ct] += v * v;
        }
      }
    }
  }
#pragma unroll
  for (int ct = 0; ct < 4; ++ct) {
    int col = ct * 16 + lrow;
    atomicAdd(&s_sum[col], lsum[ct]);
    atomicAdd(&s_sq[col], lsq[ct]);
  }
  __syncthreads();
  if (threadIdx.x < 64) {
    atomicAdd(&stats[threadIdx.x],      s_sum[threadIdx.x]);
    atomicAdd(&stats[64 + threadIdx.x], s_sq[threadIdx.x]);
  }
}

// ---------------------------------------------------------------------------
// BN3 + ReLU in-place on d_out (f32)
// ---------------------------------------------------------------------------
__global__ void k_bn3(float* __restrict__ o, const float* __restrict__ ss, int total) {
  int i = blockIdx.x * 256 + threadIdx.x;
  if (i < total) {
    int c = i & 63;
    o[i] = fmaxf(fmaf(o[i], ss[c], ss[64 + c]), 0.0f);
  }
}

// ---------------------------------------------------------------------------
extern "C" void kernel_launch(void* const* d_in, const int* in_sizes, int n_in,
                              void* d_out, int out_size, void* d_ws, size_t ws_size,
                              hipStream_t stream) {
  const float* feats   = (const float*)d_in[0];
  const float* t       = (const float*)d_in[1];
  const float* w_fuse  = (const float*)d_in[2];
  const float* g1      = (const float*)d_in[3];
  const float* b1      = (const float*)d_in[4];
  const float* w_temb  = (const float*)d_in[5];
  const float* b_temb  = (const float*)d_in[6];
  const float* w3      = (const float*)d_in[7];
  const float* g2      = (const float*)d_in[8];
  const float* b2      = (const float*)d_in[9];
  const float* wup     = (const float*)d_in[10];
  const float* g3      = (const float*)d_in[11];
  const float* b3      = (const float*)d_in[12];
  const int* batch_idx = (const int*)d_in[13];
  const int* nbr       = (const int*)d_in[14];
  float* out           = (float*)d_out;

  const int N = in_sizes[0] / 128;

  char* ws = (char*)d_ws;
  size_t off = 0;
  auto alloc = [&](size_t bytes) -> void* {
    void* p = ws + off;
    off = (off + bytes + 255) & ~(size_t)255;
    return p;
  };
  float*    Y    = (float*)alloc((size_t)N * 64 * sizeof(float));   // pre-BN GEMM out (reused)
  _Float16* H    = (_Float16*)alloc((size_t)N * 64 * 2);            // f16 activations
  _Float16* H2   = (_Float16*)alloc((size_t)N * 64 * 2);
  _Float16* WFT  = (_Float16*)alloc(64 * 128 * 2);
  _Float16* W3T  = (_Float16*)alloc(27 * 4096 * 2);
  _Float16* WUPT = (_Float16*)alloc(8 * 4096 * 2);
  float*    EMB  = (float*)alloc(8 * 64 * sizeof(float));
  float*    ST   = (float*)alloc(3 * 128 * sizeof(float));          // sum/sumsq x3
  float*    SS   = (float*)alloc(3 * 128 * sizeof(float));          // scale/shift x3

  const int NE = N * 64;

  k_zero<<<2, 256, 0, stream>>>(ST, 3 * 128);
  {
    int total = 64 * 128 + 27 * 4096 + 8 * 4096;
    k_prep<<<(total + 255) / 256, 256, 0, stream>>>(w_fuse, w3, wup, WFT, W3T, WUPT);
  }
  k_emb<<<1, 512, 0, stream>>>(t, w_temb, b_temb, EMB);

  // Stage 1: fuse GEMM + BN stats
  k_gemm_fuse<<<(N + 127) / 128, 256, 0, stream>>>(feats, WFT, Y, ST, N);
  k_finalize<<<1, 64, 0, stream>>>(ST, g1, b1, SS, 1.0f / (float)N);
  k_bn1<<<(NE + 255) / 256, 256, 0, stream>>>(Y, SS, EMB, batch_idx, H, NE);

  // Stage 2: 27-offset gather-GEMM-scatter + BN stats
  k_conv<<<(N + 63) / 64, 128, 0, stream>>>(H, W3T, nbr, Y, ST + 128, N);
  k_finalize<<<1, 64, 0, stream>>>(ST + 128, g2, b2, SS + 128, 1.0f / (float)N);
  k_bn2<<<(NE + 255) / 256, 256, 0, stream>>>(Y, SS + 128, H2, NE);

  // Stage 3: transposed conv (8 children) + BN stats, then in-place BN on out
  k_upconv<<<(N + 63) / 64, 128, 0, stream>>>(H2, WUPT, out, ST + 256, N);
  k_finalize<<<1, 64, 0, stream>>>(ST + 256, g3, b3, SS + 256, 1.0f / (float)(8 * N));
  k_bn3<<<(out_size + 255) / 256, 256, 0, stream>>>(out, SS + 256, out_size);
}
